// GraphAT_24910810316938
// MI455X (gfx1250) — matrix operand order repaired and checked by reading, not compiled, not run on hardware
//
#include <hip/hip_runtime.h>
#include <stdint.h>

typedef __attribute__((ext_vector_type(16))) _Float16 v16h;
typedef __attribute__((ext_vector_type(8)))  float    v8f;

// ---------------- helpers: ordered float <-> uint for atomicMax ----------------
__device__ __forceinline__ unsigned gat_fmap(float f) {
  unsigned u = __float_as_uint(f);
  return (u & 0x80000000u) ? ~u : (u | 0x80000000u);
}
__device__ __forceinline__ float gat_funmap(unsigned u) {
  return (u & 0x80000000u) ? __uint_as_float(u & 0x7FFFFFFFu) : __uint_as_float(~u);
}

// ---------------- zero fill ----------------
__global__ void gat_zero(float* p, size_t n) {
  size_t i = (size_t)blockIdx.x * blockDim.x + threadIdx.x;
  if (i < n) p[i] = 0.0f;
}

// ---------------- fp32 -> fp16 convert ----------------
__global__ void gat_cvt_f16(const float* __restrict__ in, _Float16* __restrict__ out, size_t n) {
  size_t i = (size_t)blockIdx.x * blockDim.x + threadIdx.x;
  if (i < n) out[i] = (_Float16)in[i];
}

// ---------------- pack W [K,Nn] (row-major f32) into wave32 B fragments ----------------
// Fragment dword index: ((tn*ksteps + ks)*32 + lane)*8 + v
//   column n = tn*16 + (lane&15);  k = ks*32 + (lane>>4)*16 + 2v  (pair k, k+1)
__global__ void gat_pack_w(const float* __restrict__ W, uint32_t* __restrict__ out,
                           int K, int Nn) {
  int t = blockIdx.x * blockDim.x + threadIdx.x;
  int ksteps = K >> 5, tiles_n = Nn >> 4;
  int total = tiles_n * ksteps * 32 * 8;
  if (t >= total) return;
  int v     = t & 7;
  int rest  = t >> 3;
  int lane  = rest & 31;
  int rest2 = rest >> 5;
  int ks    = rest2 % ksteps;
  int tn    = rest2 / ksteps;
  int n = tn * 16 + (lane & 15);
  int k = ks * 32 + (lane >> 4) * 16 + 2 * v;
  union { _Float16 h[2]; uint32_t u; } p;
  p.h[0] = (_Float16)W[(size_t)k * Nn + n];
  p.h[1] = (_Float16)W[(size_t)(k + 1) * Nn + n];
  out[t] = p.u;
}

// ---------------- WMMA GEMM: C[M,Nn] = A[M,K](f16 row-major) x Bpacked ----------------
// one wave per 16x16 output tile; K multiple of 32; M multiple of 16; Nn multiple of 16
union GatFrag { uint4 u[2]; v16h h; };

__global__ void gat_gemm_wmma(const _Float16* __restrict__ A,
                              const uint32_t* __restrict__ Bp,
                              float* __restrict__ C,
                              int M, int K, int Nn) {
  const int tiles_n = Nn >> 4;
  const int tiles_m = M >> 4;
  const int ksteps  = K >> 5;
  int wave = (int)((blockIdx.x * (size_t)blockDim.x + threadIdx.x) >> 5);
  int lane = threadIdx.x & 31;
  int tm = wave / tiles_n, tn = wave % tiles_n;
  if (tm >= tiles_m) return;                 // wave-uniform exit: EXEC stays all-ones
  int half = lane >> 4, l15 = lane & 15;

  const _Float16* arow = A + (size_t)(tm * 16 + l15) * K + half * 16;
  const uint4* b4 = (const uint4*)Bp + ((size_t)tn * ksteps * 32 + lane) * 2;

  v8f acc = {};
  for (int ks = 0; ks < ksteps; ++ks) {
    GatFrag a, b;
    a.u[0] = *(const uint4*)(arow + ks * 32);        // VGPR 0-3: K = ks*32 + half*16 + [0..7]
    a.u[1] = *(const uint4*)(arow + ks * 32 + 8);    // VGPR 4-7: K = ks*32 + half*16 + 8 + [0..7]
    b.u[0] = b4[(size_t)ks * 64];
    b.u[1] = b4[(size_t)ks * 64 + 1];
    acc = __builtin_amdgcn_wmma_f32_16x16x32_f16(false, a.h, false, b.h,
                                                 (short)0, acc, false, false);
  }
  float* crow = C + (size_t)(tm * 16 + half * 8) * Nn + tn * 16 + l15;
#pragma unroll
  for (int r = 0; r < 8; ++r) crow[(size_t)r * Nn] = acc[r];
}

// ---------------- attention coefficients: a_s/a_d [N,3] ----------------
__global__ void gat_att(const float* __restrict__ h, const float* __restrict__ att_s,
                        const float* __restrict__ att_d, float* __restrict__ as_,
                        float* __restrict__ ad_, int N, int C) {
  int t = blockIdx.x * blockDim.x + threadIdx.x;
  if (t >= N * 3) return;
  int n = t / 3, hh = t % 3;
  const float* hp = h + (size_t)n * 3 * C + (size_t)hh * C;
  const float* sp = att_s + hh * C;
  const float* dp = att_d + hh * C;
  float s = 0.f, d = 0.f;
  for (int c = 0; c < C; ++c) { float v = hp[c]; s += v * sp[c]; d += v * dp[c]; }
  as_[t] = s; ad_[t] = d;
}

// ---------------- per-edge logit + segment max ----------------
__global__ void gat_edge_logit(const int* __restrict__ ei, int E, int N,
                               const float* __restrict__ as_, const float* __restrict__ ad_,
                               float* __restrict__ ebuf, unsigned* __restrict__ mu) {
  int t = blockIdx.x * blockDim.x + threadIdx.x;
  int Etot = E + N;
  if (t >= Etot * 3) return;
  int e = t / 3, hh = t % 3;
  int src = (e < E) ? ei[e]     : (e - E);
  int dst = (e < E) ? ei[E + e] : (e - E);
  float v = as_[src * 3 + hh] + ad_[dst * 3 + hh];
  v = v > 0.f ? v : 0.2f * v;               // leaky_relu slope 0.2
  ebuf[t] = v;
  atomicMax(&mu[dst * 3 + hh], gat_fmap(v));
}

// ---------------- per-edge exp + segment sum ----------------
__global__ void gat_edge_exp(const int* __restrict__ ei, int E, int N,
                             float* __restrict__ ebuf, const unsigned* __restrict__ mu,
                             float* __restrict__ denom) {
  int t = blockIdx.x * blockDim.x + threadIdx.x;
  int Etot = E + N;
  if (t >= Etot * 3) return;
  int e = t / 3, hh = t % 3;
  int dst = (e < E) ? ei[E + e] : (e - E);
  float m = gat_funmap(mu[dst * 3 + hh]);
  float ex = __expf(ebuf[t] - m);
  ebuf[t] = ex;
  atomicAdd(&denom[dst * 3 + hh], ex);
}

// ---------------- scatter: out[dst] += h[src]*alpha (wave per edge) ----------------
__global__ void gat_scatter(const int* __restrict__ ei, int E, int N,
                            const float* __restrict__ h, const float* __restrict__ ebuf,
                            const float* __restrict__ denom, float* __restrict__ out, int C) {
  int wave = (int)((blockIdx.x * (size_t)blockDim.x + threadIdx.x) >> 5);
  int lane = threadIdx.x & 31;
  int Etot = E + N;
  if (wave >= Etot) return;
  int e = wave;
  int src = (e < E) ? ei[e]     : (e - E);
  int dst = (e < E) ? ei[E + e] : (e - E);
  int HC = 3 * C;
  for (int i = lane; i < HC; i += 32) {
    int hh = i / C;
    float alpha = ebuf[e * 3 + hh] / (denom[dst * 3 + hh] + 1e-16f);
    atomicAdd(&out[(size_t)dst * HC + i], h[(size_t)src * HC + i] * alpha);
  }
}

// ---------------- head mean + bias + batchnorm + leaky -> hid (f16) ----------------
__global__ void gat_bn(const float* __restrict__ out1, const float* __restrict__ bias1,
                       const float* __restrict__ gamma, const float* __restrict__ beta,
                       const float* __restrict__ mean, const float* __restrict__ var,
                       _Float16* __restrict__ hidh, int N) {
  int t = blockIdx.x * blockDim.x + threadIdx.x;
  if (t >= N * 64) return;
  int n = t >> 6, c = t & 63;
  const float* p = out1 + (size_t)n * 192;
  float s = (p[c] + p[64 + c] + p[128 + c]) * (1.f / 3.f) + bias1[c];
  float v = (s - mean[c]) * rsqrtf(var[c] + 1e-5f) * gamma[c] + beta[c];
  v = v > 0.f ? v : 0.01f * v;              // F.leaky_relu default slope
  hidh[t] = (_Float16)v;
}

// ---------------- final: head mean + bias + log_softmax + masked NLL ----------------
__global__ void gat_final(const float* __restrict__ out2, const float* __restrict__ bias2,
                          const int* __restrict__ y, const unsigned char* __restrict__ mask,
                          float* __restrict__ logp, float* __restrict__ lossacc, int N) {
  int n = blockIdx.x * blockDim.x + threadIdx.x;
  if (n >= N) return;
  const float* p = out2 + (size_t)n * 48;
  float o[16];
  float mx = -3.4e38f;
#pragma unroll
  for (int c = 0; c < 16; ++c) {
    o[c] = (p[c] + p[16 + c] + p[32 + c]) * (1.f / 3.f) + bias2[c];
    mx = fmaxf(mx, o[c]);
  }
  float s = 0.f;
#pragma unroll
  for (int c = 0; c < 16; ++c) s += __expf(o[c] - mx);
  float lz = mx + __logf(s);
#pragma unroll
  for (int c = 0; c < 16; ++c) logp[(size_t)n * 16 + c] = o[c] - lz;
  if (mask[n]) {
    atomicAdd(&lossacc[0], -(o[y[n]] - lz));
    atomicAdd(&lossacc[1], 1.0f);
  }
}

__global__ void gat_loss_fin(const float* __restrict__ lossacc, float* __restrict__ out, int N) {
  out[(size_t)N * 16] = lossacc[0] / lossacc[1];
}

// ==========================================================================
extern "C" void kernel_launch(void* const* d_in, const int* in_sizes, int n_in,
                              void* d_out, int out_size, void* d_ws, size_t ws_size,
                              hipStream_t stream) {
  (void)n_in; (void)out_size; (void)ws_size;
  const float* x      = (const float*)d_in[0];
  const int*   ei     = (const int*)d_in[1];
  const int*   y      = (const int*)d_in[2];
  const unsigned char* mask = (const unsigned char*)d_in[3];
  const float* W1     = (const float*)d_in[4];
  const float* att_s1 = (const float*)d_in[5];
  const float* att_d1 = (const float*)d_in[6];
  const float* bias1  = (const float*)d_in[7];
  const float* gamma  = (const float*)d_in[8];
  const float* beta   = (const float*)d_in[9];
  const float* mean   = (const float*)d_in[10];
  const float* var    = (const float*)d_in[11];
  const float* W2     = (const float*)d_in[12];
  const float* att_s2 = (const float*)d_in[13];
  const float* att_d2 = (const float*)d_in[14];
  const float* bias2  = (const float*)d_in[15];

  const int IN_C = 256;
  int N    = in_sizes[0] / IN_C;   // 50000
  int E    = in_sizes[1] / 2;      // 800000
  int Etot = E + N;

  // -------- carve workspace --------
  char* ws = (char*)d_ws;
  size_t off = 0;
  auto carve = [&](size_t bytes) -> void* {
    void* p = ws + off;
    off += (bytes + 255) & ~(size_t)255;
    return p;
  };
  _Float16* xh   = (_Float16*)carve((size_t)N * 256 * 2);
  uint32_t* W1p  = (uint32_t*)carve((size_t)12 * 8 * 32 * 8 * 4);  // 192/16 tiles, 256/32 ksteps
  uint32_t* W2p  = (uint32_t*)carve((size_t)3 * 2 * 32 * 8 * 4);   // 48/16 tiles, 64/32 ksteps
  float*    h1   = (float*)carve((size_t)N * 192 * 4);
  float*    as1  = (float*)carve((size_t)N * 3 * 4);
  float*    ad1  = (float*)carve((size_t)N * 3 * 4);
  float*    as2  = (float*)carve((size_t)N * 3 * 4);
  float*    ad2  = (float*)carve((size_t)N * 3 * 4);
  float*    h2   = (float*)carve((size_t)N * 48 * 4);
  float*    ebuf = (float*)carve((size_t)Etot * 3 * 4);
  _Float16* hidh = (_Float16*)carve((size_t)N * 64 * 2);
  // zero-initialized region (contiguous)
  char* z0 = ws + off;
  unsigned* mu1  = (unsigned*)carve((size_t)N * 3 * 4);
  float*    dn1  = (float*)carve((size_t)N * 3 * 4);
  unsigned* mu2  = (unsigned*)carve((size_t)N * 3 * 4);
  float*    dn2  = (float*)carve((size_t)N * 3 * 4);
  float*    out1 = (float*)carve((size_t)N * 192 * 4);
  float*    out2 = (float*)carve((size_t)N * 48 * 4);
  float*    lossacc = (float*)carve(2 * 4);
  size_t zfloats = (size_t)((ws + off) - z0) / 4;

  const int B = 256;
  // zero init
  gat_zero<<<(int)((zfloats + B - 1) / B), B, 0, stream>>>((float*)z0, zfloats);
  // convert x to f16
  {
    size_t n = (size_t)N * 256;
    gat_cvt_f16<<<(int)((n + B - 1) / B), B, 0, stream>>>(x, xh, n);
  }
  // pack weights into WMMA B fragments
  gat_pack_w<<<(12 * 8 * 32 * 8 + B - 1) / B, B, 0, stream>>>(W1, W1p, 256, 192);
  gat_pack_w<<<(3 * 2 * 32 * 8 + B - 1) / B, B, 0, stream>>>(W2, W2p, 64, 48);

  // ---- layer 1 ----
  {
    int waves = (N / 16) * (192 / 16);
    gat_gemm_wmma<<<(waves * 32 + B - 1) / B, B, 0, stream>>>(xh, W1p, h1, N, 256, 192);
  }
  gat_att<<<(N * 3 + B - 1) / B, B, 0, stream>>>(h1, att_s1, att_d1, as1, ad1, N, 64);
  gat_edge_logit<<<(Etot * 3 + B - 1) / B, B, 0, stream>>>(ei, E, N, as1, ad1, ebuf, mu1);
  gat_edge_exp<<<(Etot * 3 + B - 1) / B, B, 0, stream>>>(ei, E, N, ebuf, mu1, dn1);
  gat_scatter<<<(int)(((size_t)Etot * 32 + B - 1) / B), B, 0, stream>>>(ei, E, N, h1, ebuf, dn1, out1, 64);
  gat_bn<<<(N * 64 + B - 1) / B, B, 0, stream>>>(out1, bias1, gamma, beta, mean, var, hidh, N);

  // ---- layer 2 ----
  {
    int waves = (N / 16) * (48 / 16);
    gat_gemm_wmma<<<(waves * 32 + B - 1) / B, B, 0, stream>>>(hidh, W2p, h2, N, 64, 48);
  }
  gat_att<<<(N * 3 + B - 1) / B, B, 0, stream>>>(h2, att_s2, att_d2, as2, ad2, N, 16);
  gat_edge_logit<<<(Etot * 3 + B - 1) / B, B, 0, stream>>>(ei, E, N, as2, ad2, ebuf, mu2);
  gat_edge_exp<<<(Etot * 3 + B - 1) / B, B, 0, stream>>>(ei, E, N, ebuf, mu2, dn2);
  gat_scatter<<<(int)(((size_t)Etot * 32 + B - 1) / B), B, 0, stream>>>(ei, E, N, h2, ebuf, dn2, out2, 16);

  // ---- output ----
  gat_final<<<(N + B - 1) / B, B, 0, stream>>>(out2, bias2, y, mask, (float*)d_out, lossacc, N);
  gat_loss_fin<<<1, 1, 0, stream>>>(lossacc, (float*)d_out, N);
}
